// TokenDistributionRouter_71502615544492
// MI455X (gfx1250) — compile-verified, matrix operand order checked by reading, not run on hardware
//
#include <hip/hip_runtime.h>
#include <hip/hip_bf16.h>
#include <math.h>

typedef __attribute__((ext_vector_type(16))) _Float16 v16h;
typedef __attribute__((ext_vector_type(8)))  _Float16 v8h;
typedef __attribute__((ext_vector_type(8)))  float    v8f;

#define N_   8192
#define D_   4096
#define L_   256
#define M_   64
#define TWO_L 512

// ---- workspace layout (float offsets) ----
#define MU_OFF    0                    // N*512 f32
#define Z_OFF     4194304              // N*256 f32
#define WV_OFF    6291456              // N*256 f16 (occupies 1048576 floats)
#define S_OFF     7340032              // 256 f32
#define KL_OFF    7340288              // 1 f32 (padded)
#define WKT_OFF   7340544              // 64*256 f32
#define WENCP_OFF 7356928              // 4096*512 f16 packed (1048576 floats)
#define WOUTP_OFF 8405504              // 256*4096 f16 packed (524288 floats)

// ---- output layout (float offsets) ----
#define RW_OUT   0          // N*2
#define AUX_OUT  16384      // 1
#define IDX_OUT  16385      // N*2 (indices stored as float)
#define SC_OUT   32769      // N*64
#define ZD_OUT   557057     // N*4096

// 16-bit WMMA fragment K mapping (ISA 7.12.2: 16-bit A 16x32 table)
__device__ __forceinline__ int kmap(int e, int laneHi) {
    return e + 8 * ((e >> 3) + laneHi);
}

__device__ __forceinline__ v16h frag_from_2x8(v8h lo, v8h hi8) {
    return __builtin_shufflevector(lo, hi8,
        0, 1, 2, 3, 4, 5, 6, 7, 8, 9, 10, 11, 12, 13, 14, 15);
}

// fast reciprocal (v_rcp_f32) — plenty of precision for f16 WMMA inputs
__device__ __forceinline__ float fast_rcp(float x) {
    return __builtin_amdgcn_rcpf(x);
}

// ============================================================
// Kernel 0: repack f32 weight [K x Ncols] -> f16 fragment-major
//   dst[(kt*Ncols + n)*32 + (k&31)] = (f16)src[k*Ncols + n]
// so a lane's 8 consecutive k-values are one 16B global_load_b128.
// ============================================================
__global__ void pack_w_kernel(const float* __restrict__ Wsrc,
                              _Float16* __restrict__ Wdst,
                              int total, int Ncols)
{
    int i = blockIdx.x * blockDim.x + threadIdx.x;
    if (i >= total) return;
    int k = i / Ncols, n = i - k * Ncols;
    int kt = k >> 5, kr = k & 31;
    Wdst[((size_t)kt * Ncols + n) * 32 + kr] = (_Float16)Wsrc[i];
}

// ============================================================
// Kernel 1: fused LayerNorm + SiLU + encoder GEMM (WMMA f16)
//   mu_logvar[row, 0:512] = silu(LN(x[row,:])) @ W_enc + b_enc
//   block = 256 thr (8 waves), 16 rows x 512 cols per block
//   A tile double-buffered: single barrier per k-step
// ============================================================
__global__ __launch_bounds__(256) void enc_gemm_kernel(
    const float* __restrict__ x,
    const float* __restrict__ gamma, const float* __restrict__ beta,
    const _Float16* __restrict__ WencP, const float* __restrict__ benc,
    float* __restrict__ mu_logvar)
{
    __shared__ _Float16 As[2][16][40];   // double-buffered 16x32 A tile
    __shared__ float meanS[16], rstdS[16];

    const int row0 = blockIdx.x * 16;
    const int tid  = threadIdx.x;
    const int lane = tid & 31;
    const int wave = tid >> 5;

    // ---- prologue: per-row mean / rstd (wave w handles rows 2w, 2w+1) ----
    for (int i = 0; i < 2; ++i) {
        int r = wave * 2 + i;
        const float4* xr = (const float4*)(x + (size_t)(row0 + r) * D_);
        float s = 0.f, ss = 0.f;
        for (int j = lane; j < D_ / 4; j += 32) {
            float4 v = xr[j];
            s  += v.x + v.y + v.z + v.w;
            ss += v.x*v.x + v.y*v.y + v.z*v.z + v.w*v.w;
        }
        for (int off = 16; off; off >>= 1) {
            s  += __shfl_xor(s,  off, 32);
            ss += __shfl_xor(ss, off, 32);
        }
        if (lane == 0) {
            float m   = s / (float)D_;
            float var = ss / (float)D_ - m * m;
            meanS[r] = m;
            rstdS[r] = rsqrtf(var + 1e-5f);
        }
    }
    __syncthreads();

    const int wc0 = wave * 64;           // this wave's 64 output cols
    const int m   = lane & 15;
    const int hi  = lane >> 4;
    v8f acc[4] = {};

    // staging coords for this thread (2 consecutive elems)
    const int sr = (tid * 2) >> 5;
    const int sc = (tid * 2) & 31;
    const float* xrow = x + (size_t)(row0 + sr) * D_;

    for (int kk = 0; kk < D_; kk += 32) {
        const int pb = (kk >> 5) & 1;
        // stage A tile: 16x32 of h = silu(LN(x)), float2 per thread
        {
            float2 xv = *(const float2*)(xrow + kk + sc);
            float2 gv = *(const float2*)(gamma + kk + sc);
            float2 bv = *(const float2*)(beta  + kk + sc);
            float mn = meanS[sr], rs = rstdS[sr];
            float h0 = (xv.x - mn) * rs * gv.x + bv.x;
            float h1 = (xv.y - mn) * rs * gv.y + bv.y;
            h0 = h0 * fast_rcp(1.f + __expf(-h0));   // SiLU via v_rcp_f32
            h1 = h1 * fast_rcp(1.f + __expf(-h1));
            As[pb][sr][sc]     = (_Float16)h0;
            As[pb][sr][sc + 1] = (_Float16)h1;
            if (kk + 256 < D_)           // stream prefetch 8 tiles ahead
                __builtin_prefetch(xrow + kk + 256 + sc, 0, 0);
        }
        __syncthreads();                 // single barrier per k-step

        v16h a;
        #pragma unroll
        for (int e = 0; e < 16; ++e) a[e] = As[pb][m][kmap(e, hi)];

        const int kt = kk >> 5;
        #pragma unroll
        for (int t = 0; t < 4; ++t) {
            const int n = wc0 + t * 16 + m;     // B column owned by this lane
            const v8h* bp = (const v8h*)(WencP + ((size_t)kt * TWO_L + n) * 32);
            v16h b = frag_from_2x8(bp[hi], bp[2 + hi]);
            acc[t] = __builtin_amdgcn_wmma_f32_16x16x32_f16(
                false, a, false, b, (short)0, acc[t], false, false);
        }
    }

    // epilogue: + bias, store f32
    #pragma unroll
    for (int t = 0; t < 4; ++t) {
        int col = wc0 + t * 16 + m;
        float bias = benc[col];
        #pragma unroll
        for (int r = 0; r < 8; ++r) {
            int row = row0 + r + hi * 8;
            mu_logvar[(size_t)row * TWO_L + col] = acc[t][r] + bias;
        }
    }
}

// ============================================================
// Kernel 2: zero global accumulators (s, kl, wkt): 16896 floats
// ============================================================
__global__ void zero_kernel(float* __restrict__ p, int n)
{
    int i = blockIdx.x * blockDim.x + threadIdx.x;
    if (i < n) p[i] = 0.f;
}

// ============================================================
// Kernel 3: per-row router. 64 rows/block, 8 rows/wave.
// ============================================================
__global__ __launch_bounds__(256) void router_kernel(
    const float* __restrict__ mu_logvar, const float* __restrict__ eps,
    const float* __restrict__ keys,
    float* __restrict__ z, _Float16* __restrict__ wv,
    float* __restrict__ s_acc, float* __restrict__ kl_acc,
    float* __restrict__ wkt_acc,
    float* __restrict__ rw_out, float* __restrict__ idx_out,
    float* __restrict__ scores_out)
{
    __shared__ float wkt[M_ * L_];       // 64 KB partial target accumulation
    __shared__ float sP[L_];
    __shared__ float muB[8][L_];
    __shared__ float pB[8][M_];
    __shared__ float klW[8];

    const int tid  = threadIdx.x;
    const int lane = tid & 31;
    const int wave = tid >> 5;
    const int row0 = blockIdx.x * 64;

    for (int i = tid; i < M_ * L_; i += 256) wkt[i] = 0.f;
    if (tid < L_) sP[tid] = 0.f;
    __syncthreads();

    float klLane = 0.f;

    for (int rr = 0; rr < 8; ++rr) {
        const int row = row0 + wave * 8 + rr;

        // ---- load mu/logvar, reparameterize, KL + norm partials ----
        float muv[8], zv[8];
        float nsq = 0.f;
        #pragma unroll
        for (int jj = 0; jj < 8; ++jj) {
            int j = lane + 32 * jj;
            float mu = mu_logvar[(size_t)row * TWO_L + j];
            float lv = mu_logvar[(size_t)row * TWO_L + L_ + j];
            float ev = eps[(size_t)row * L_ + j];
            float zz = mu + ev * __expf(0.5f * lv);
            muv[jj] = mu; zv[jj] = zz;
            z[(size_t)row * L_ + j] = zz;
            klLane += 1.f + lv - mu * mu - __expf(lv);
            nsq += mu * mu;
            muB[wave][j] = mu;
        }
        float red = nsq;
        for (int off = 16; off; off >>= 1) red += __shfl_xor(red, off, 32);
        float inv = fast_rcp(fmaxf(sqrtf(red), 1e-12f));
        #pragma unroll
        for (int jj = 0; jj < 8; ++jj)
            atomicAdd(&sP[lane + 32 * jj], muv[jj] * inv);

        // ---- scores: lane handles m = lane, lane+32 ----
        float sc[2];
        #pragma unroll
        for (int mi = 0; mi < 2; ++mi) {
            int mm = lane + 32 * mi;
            float a = 0.f;
            for (int j = 0; j < L_; ++j) a += muB[wave][j] * keys[mm * L_ + j];
            sc[mi] = a;
            scores_out[(size_t)row * M_ + mm] = a;
        }

        // ---- softmax over 64 ----
        float mx = fmaxf(sc[0], sc[1]);
        for (int off = 16; off; off >>= 1) mx = fmaxf(mx, __shfl_xor(mx, off, 32));
        float e0 = __expf(sc[0] - mx), e1 = __expf(sc[1] - mx);
        float Zs = e0 + e1;
        for (int off = 16; off; off >>= 1) Zs += __shfl_xor(Zs, off, 32);
        float rZs = fast_rcp(Zs);
        pB[wave][lane]      = e0 * rZs;
        pB[wave][lane + 32] = e1 * rZs;

        // ---- weighted_v = softmax @ keys, stored f16 ----
        #pragma unroll
        for (int jj = 0; jj < 8; ++jj) {
            int j = lane + 32 * jj;
            float a = 0.f;
            for (int mm = 0; mm < M_; ++mm) a += pB[wave][mm] * keys[mm * L_ + j];
            wv[(size_t)row * L_ + j] = (_Float16)a;
        }

        // ---- top-2 (value,index) wave reduction ----
        float v0 = sc[0]; int i0 = lane;
        float v1 = sc[1]; int i1 = lane + 32;
        if (v1 > v0) { float tv = v0; v0 = v1; v1 = tv; int ti = i0; i0 = i1; i1 = ti; }
        for (int off = 16; off; off >>= 1) {
            float ov0 = __shfl_xor(v0, off, 32);
            int   oi0 = __shfl_xor(i0, off, 32);
            float ov1 = __shfl_xor(v1, off, 32);
            int   oi1 = __shfl_xor(i1, off, 32);
            if (ov0 > v0) {
                if (v0 > ov1) { v1 = v0; i1 = i0; } else { v1 = ov1; i1 = oi1; }
                v0 = ov0; i0 = oi0;
            } else if (ov0 > v1) { v1 = ov0; i1 = oi0; }
        }
        if (lane == 0) {
            float w0 = fast_rcp(1.f + __expf(v1 - v0));  // softmax of top-2
            rw_out[(size_t)row * 2 + 0] = w0;
            rw_out[(size_t)row * 2 + 1] = 1.f - w0;
            idx_out[(size_t)row * 2 + 0] = (float)i0;
            idx_out[(size_t)row * 2 + 1] = (float)i1;
        }

        // ---- last_routing = softmax(scatter(topk)), then wkt += lr^T z ----
        float mx2 = fmaxf(v0, 0.f);
        float rZ2 = fast_rcp(__expf(v0 - mx2) + __expf(v1 - mx2)
                             + (float)(M_ - 2) * __expf(-mx2));
        #pragma unroll
        for (int mi = 0; mi < 2; ++mi) {
            int mm = lane + 32 * mi;
            float rs = (mm == i0) ? v0 : ((mm == i1) ? v1 : 0.f);
            pB[wave][mm] = __expf(rs - mx2) * rZ2;
        }
        #pragma unroll
        for (int jj = 0; jj < 8; ++jj) {
            int j = lane + 32 * jj;
            float zj = zv[jj];
            for (int mm = 0; mm < M_; ++mm)
                atomicAdd(&wkt[mm * L_ + j], pB[wave][mm] * zj);
        }
    }

    // ---- block epilogue: flush partials ----
    for (int off = 16; off; off >>= 1) klLane += __shfl_xor(klLane, off, 32);
    if (lane == 0) klW[wave] = klLane;
    __syncthreads();
    if (tid == 0) {
        float k = 0.f;
        for (int w = 0; w < 8; ++w) k += klW[w];
        atomicAdd(kl_acc, -0.5f * k);
    }
    if (tid < L_) atomicAdd(&s_acc[tid], sP[tid]);
    for (int i = tid; i < M_ * L_; i += 256) atomicAdd(&wkt_acc[i], wkt[i]);
}

// ============================================================
// Kernel 4: decoder GEMM (WMMA f16): z_decoded = wv @ W_out + b_out
//   block = 256 thr, 16 rows x 512 cols; K = 256
// ============================================================
__global__ __launch_bounds__(256) void dec_gemm_kernel(
    const _Float16* __restrict__ wv,
    const _Float16* __restrict__ WoutP, const float* __restrict__ bout,
    float* __restrict__ zdec)
{
    const int tilesPerRow = D_ / 512;                 // 8
    const int row0  = (blockIdx.x / tilesPerRow) * 16;
    const int wc00  = (blockIdx.x % tilesPerRow) * 512;
    const int tid   = threadIdx.x;
    const int lane  = tid & 31;
    const int wave  = tid >> 5;
    const int wc0   = wc00 + wave * 64;
    const int m     = lane & 15;
    const int hi    = lane >> 4;

    v8f acc[4] = {};

    for (int kk = 0; kk < L_; kk += 32) {
        // A fragment: wv row-major f16, two 16B loads per lane
        const v8h* ap = (const v8h*)(wv + (size_t)(row0 + m) * L_ + kk);
        v16h a = frag_from_2x8(ap[hi], ap[2 + hi]);

        const int kt = kk >> 5;
        #pragma unroll
        for (int t = 0; t < 4; ++t) {
            const int n = wc0 + t * 16 + m;
            const v8h* bp = (const v8h*)(WoutP + ((size_t)kt * D_ + n) * 32);
            v16h b = frag_from_2x8(bp[hi], bp[2 + hi]);
            acc[t] = __builtin_amdgcn_wmma_f32_16x16x32_f16(
                false, a, false, b, (short)0, acc[t], false, false);
        }
    }

    #pragma unroll
    for (int t = 0; t < 4; ++t) {
        int col = wc0 + t * 16 + m;
        float bias = bout[col];
        #pragma unroll
        for (int r = 0; r < 8; ++r) {
            int row = row0 + r + hi * 8;
            zdec[(size_t)row * D_ + col] = acc[t][r] + bias;
        }
    }
}

// ============================================================
// Kernel 5: scalar finalize -> aux_loss
// ============================================================
__global__ __launch_bounds__(256) void finalize_kernel(
    const float* __restrict__ s_acc, const float* __restrict__ kl_acc,
    const float* __restrict__ wkt_acc, const float* __restrict__ keys,
    float* __restrict__ aux_out)
{
    __shared__ float red[256];
    __shared__ float tvec[L_];
    __shared__ float knorm[M_];
    const int tid = threadIdx.x;

    if (tid < M_) {
        float ns = 0.f;
        for (int j = 0; j < L_; ++j) { float k = keys[tid * L_ + j]; ns += k * k; }
        knorm[tid] = 1.f / fmaxf(sqrtf(ns), 1e-12f);
    }
    __syncthreads();
    if (tid < L_) {
        float tv = 0.f;
        for (int mm = 0; mm < M_; ++mm) tv += keys[mm * L_ + tid] * knorm[mm];
        tvec[tid] = tv;
    }
    __syncthreads();

    // reduce 1: |s|^2
    red[tid] = (tid < L_) ? s_acc[tid] * s_acc[tid] : 0.f;
    __syncthreads();
    for (int st = 128; st; st >>= 1) { if (tid < st) red[tid] += red[tid + st]; __syncthreads(); }
    float ss = red[0];
    __syncthreads();

    // reduce 2: |sum n_k|^2
    red[tid] = (tid < L_) ? tvec[tid] * tvec[tid] : 0.f;
    __syncthreads();
    for (int st = 128; st; st >>= 1) { if (tid < st) red[tid] += red[tid + st]; __syncthreads(); }
    float tt = red[0];
    __syncthreads();

    // reduce 3: sim_loss numerator
    float sl = 0.f;
    for (int i = tid; i < M_ * L_; i += 256) sl += fabsf(keys[i] - wkt_acc[i]);
    red[tid] = sl;
    __syncthreads();
    for (int st = 128; st; st >>= 1) { if (tid < st) red[tid] += red[tid + st]; __syncthreads(); }

    if (tid == 0) {
        const float Nf = (float)N_;
        float div_mu   = (ss - Nf) / (Nf * (Nf - 1.f));
        float div_keys = (tt - (float)M_) / (float)(M_ * (M_ - 1));
        float div_loss = 0.1f * div_mu + 0.1f * div_keys;   // LAMBDA_DIV
        float kl       = kl_acc[0] / Nf;
        float sim_loss = red[0] / (float)(M_ * L_);
        aux_out[0] = 0.1f * div_loss + 0.1f * kl + 0.1f * sim_loss;
    }
}

// ============================================================
extern "C" void kernel_launch(void* const* d_in, const int* in_sizes, int n_in,
                              void* d_out, int out_size, void* d_ws, size_t ws_size,
                              hipStream_t stream)
{
    const float* x     = (const float*)d_in[0];
    const float* eps   = (const float*)d_in[1];
    const float* gamma = (const float*)d_in[2];
    const float* beta  = (const float*)d_in[3];
    const float* Wenc  = (const float*)d_in[4];
    const float* benc  = (const float*)d_in[5];
    const float* Wout  = (const float*)d_in[6];
    const float* bout  = (const float*)d_in[7];
    const float* keys  = (const float*)d_in[8];

    float* out = (float*)d_out;
    float* W   = (float*)d_ws;

    float*     mu_lv  = W + MU_OFF;
    float*     zbuf   = W + Z_OFF;
    _Float16*  wv     = (_Float16*)(W + WV_OFF);
    float*     s_acc  = W + S_OFF;
    float*     kl_acc = W + KL_OFF;
    float*     wktA   = W + WKT_OFF;
    _Float16*  WencP  = (_Float16*)(W + WENCP_OFF);
    _Float16*  WoutP  = (_Float16*)(W + WOUTP_OFF);

    // 0) repack weights to f16 fragment-major layout
    pack_w_kernel<<<(D_ * TWO_L + 255) / 256, 256, 0, stream>>>(Wenc, WencP, D_ * TWO_L, TWO_L);
    pack_w_kernel<<<(L_ * D_ + 255) / 256, 256, 0, stream>>>(Wout, WoutP, L_ * D_, D_);

    // 1) fused LN+SiLU encoder GEMM
    enc_gemm_kernel<<<N_ / 16, 256, 0, stream>>>(x, gamma, beta, WencP, benc, mu_lv);

    // 2) zero accumulators (s[256] + kl pad[256] + wkt[16384] contiguous)
    zero_kernel<<<(16896 + 255) / 256, 256, 0, stream>>>(s_acc, 16896);

    // 3) router
    router_kernel<<<N_ / 64, 256, 0, stream>>>(
        mu_lv, eps, keys, zbuf, wv, s_acc, kl_acc, wktA,
        out + RW_OUT, out + IDX_OUT, out + SC_OUT);

    // 4) decoder GEMM
    dec_gemm_kernel<<<(N_ / 16) * (D_ / 512), 256, 0, stream>>>(
        wv, WoutP, bout, out + ZD_OUT);

    // 5) aux loss
    finalize_kernel<<<1, 256, 0, stream>>>(s_acc, kl_acc, wktA, keys, out + AUX_OUT);
}
